// Tiresias_68221260530099
// MI455X (gfx1250) — matrix operand order repaired and checked by reading, not compiled, not run on hardware
//
#include <hip/hip_runtime.h>
#include <hip/hip_bf16.h>
#include <math.h>

// ---------------- CDNA5 WMMA types ----------------
typedef __attribute__((ext_vector_type(16))) __bf16 v16bf;
typedef __attribute__((ext_vector_type(8)))  __bf16 v8bf;
typedef __attribute__((ext_vector_type(8)))  float  v8f;
typedef __attribute__((ext_vector_type(4)))  unsigned int v4u;

#define BATCH 256
#define SEQ   128
#define HID   1024
#define GSZ   16384   // 4*HID*K
#define KCELL 4

// ---- workspace layout (bytes) ----
#define OFF_WHF 0u                    // Wh fragments, per-hb contiguous : 32 MB
#define OFF_WLF 33554432u             // Wl fragments : 2 MB
#define OFF_HB0 35651584u             // h bf16 ping : 512 KB
#define OFF_HB1 36175872u             // h bf16 pong : 512 KB
#define OFF_C   36700160u             // c state f32 (256*4*1024) : 4 MB
#define OFF_BS  40894464u             // bsum = bi + bh : 64 KB
#define ZERO_BYTES (524288u + 524288u + 4194304u)   // HB0, HB1, C (contiguous)

__device__ __forceinline__ float sigf(float x) { return 1.0f / (1.0f + __expf(-x)); }

// ---- Wh fragment swizzle, keyed by h-tile: each block's B stream is one
// contiguous 512KB region (32 chunks x 16KB).
// Layout: [hb(64)][kc(32)][ntl(16)][lane(32)][j(16)] bf16 where ntl = kk*4+g,
//   col = kk*4096 + g*1024 + hb*16 + (lane&15)
//   K   = kc*32 + 16*(lane>>4) + j                  (contiguous k order, same as the A loads)
__global__ __launch_bounds__(256) void prep_whf(const float* __restrict__ W,
                                                __bf16* __restrict__ F) {
  int tid = blockIdx.x * 256 + threadIdx.x;       // 64*32*16*32 = 1,048,576 threads
  int lane = tid & 31;
  int ntl  = (tid >> 5) & 15;
  int kc   = (tid >> 9) & 31;
  int hb   = tid >> 14;
  int kk = ntl >> 2, g = ntl & 3;
  int col = kk * 4096 + g * 1024 + hb * 16 + (lane & 15);
  int kb  = kc * 32 + ((lane >> 4) ? 16 : 0);
  __bf16* dst = F + (size_t)tid * 16;
#pragma unroll
  for (int j = 0; j < 16; ++j)
    dst[j] = (__bf16)W[(size_t)(kb + j) * GSZ + col];
}

// Generic fragment swizzle (used for Wl): [nt][kc(32)][lane(32)][j(16)]
__global__ __launch_bounds__(256) void prep_frags(const float* __restrict__ W,
                                                  __bf16* __restrict__ F,
                                                  int ncols, int nt_count) {
  int tid = blockIdx.x * 256 + threadIdx.x;
  if (tid >= nt_count * 32 * 32) return;
  int lane = tid & 31;
  int kc   = (tid >> 5) & 31;
  int nt   = tid >> 10;
  int col  = nt * 16 + (lane & 15);
  int kb   = kc * 32 + ((lane >> 4) ? 16 : 0);
  __bf16* dst = F + (size_t)tid * 16;
#pragma unroll
  for (int j = 0; j < 16; ++j)
    dst[j] = (__bf16)W[(size_t)(kb + j) * ncols + col];
}

__global__ __launch_bounds__(256) void prep_bsum(const float* __restrict__ bi,
                                                 const float* __restrict__ bh,
                                                 float* __restrict__ bs) {
  int i = blockIdx.x * 256 + threadIdx.x;
  if (i < GSZ) bs[i] = bi[i] + bh[i];
}

__global__ __launch_bounds__(256) void zero_ws(unsigned int* __restrict__ p, unsigned int nwords) {
  unsigned int i = blockIdx.x * 256u + threadIdx.x;
  if (i < nwords) p[i] = 0u;
}

// ---------------- fused LSTM step ----------------
// grid = (64 h-tiles, 4 batch-tiles), 128 threads = 4 waves.
// Each WAVE owns 16 batch rows x all 16 gate-group tiles (16 accumulators):
// in-register cell update, no gate LDS round-trip.
// B chunks (16KB each, shared by all 4 waves) are staged through a
// double-buffered LDS pipeline: global->regs->LDS overlapped with the 16
// WMMAs of the previous chunk. Cuts B L2 traffic 4x and replaces per-wmma
// L2 waits with ds_load latency.
__global__ __launch_bounds__(128) void lstm_step(
    const __bf16* __restrict__ hIn, __bf16* __restrict__ hOut,
    float* __restrict__ cSt, const __bf16* __restrict__ WhF,
    const float* __restrict__ Wi, const float* __restrict__ bsum,
    const int* __restrict__ X, int t) {
  __shared__ __bf16 bstage[2][16 * 512];           // 2 x 16KB double buffer

  const int hb    = blockIdx.x;                    // h-tile 0..63
  const int wv    = threadIdx.x >> 5;
  const int lane  = threadIdx.x & 31;
  const int lrow  = lane & 15;
  const int lhalf = lane >> 4;
  const int mbase = blockIdx.y * 64 + wv * 16;     // wave's 16 batch rows

  v8f acc[16];
#pragma unroll
  for (int i = 0; i < 16; ++i) acc[i] = (v8f){};

  const __bf16* hRow = hIn + (size_t)(mbase + lrow) * HID;
  const char*   gsrc = (const char*)(WhF + (size_t)hb * (32 * 16 * 512));
  const int     toff = threadIdx.x * 128;          // 128B per thread per 16KB chunk

  // preload chunk 0 into LDS buffer 0
  v4u st[8];
#pragma unroll
  for (int q = 0; q < 8; ++q) st[q] = *(const v4u*)(gsrc + toff + q * 16);
#pragma unroll
  for (int q = 0; q < 8; ++q)
    *(v4u*)((char*)(&bstage[0][0]) + toff + q * 16) = st[q];
  __syncthreads();

  for (int kc = 0; kc < 32; ++kc) {
    // A fragment FIRST (in-order loadcnt: waiting on A must not wait on staging)
    const int kb = kc * 32 + lhalf * 16;
    v16bf a;
    *((v8bf*)&a)     = *(const v8bf*)(hRow + kb);
    *((v8bf*)&a + 1) = *(const v8bf*)(hRow + kb + 8);

    // issue next chunk's staging loads (completion hidden behind the WMMAs)
    if (kc + 1 < 32) {
      const char* gc = gsrc + (size_t)(kc + 1) * 16384 + toff;
#pragma unroll
      for (int q = 0; q < 8; ++q) st[q] = *(const v4u*)(gc + q * 16);
    }

    // 16 WMMAs from the current LDS buffer
    const __bf16* lbuf = &bstage[kc & 1][0] + lane * 16;
#pragma unroll
    for (int q4 = 0; q4 < 4; ++q4) {
      v16bf b0, b1, b2, b3;
      *((v8bf*)&b0)     = *(const v8bf*)(lbuf + (q4 * 4 + 0) * 512);
      *((v8bf*)&b0 + 1) = *(const v8bf*)(lbuf + (q4 * 4 + 0) * 512 + 8);
      *((v8bf*)&b1)     = *(const v8bf*)(lbuf + (q4 * 4 + 1) * 512);
      *((v8bf*)&b1 + 1) = *(const v8bf*)(lbuf + (q4 * 4 + 1) * 512 + 8);
      *((v8bf*)&b2)     = *(const v8bf*)(lbuf + (q4 * 4 + 2) * 512);
      *((v8bf*)&b2 + 1) = *(const v8bf*)(lbuf + (q4 * 4 + 2) * 512 + 8);
      *((v8bf*)&b3)     = *(const v8bf*)(lbuf + (q4 * 4 + 3) * 512);
      *((v8bf*)&b3 + 1) = *(const v8bf*)(lbuf + (q4 * 4 + 3) * 512 + 8);
      acc[q4 * 4 + 0] = __builtin_amdgcn_wmma_f32_16x16x32_bf16(false, a, false, b0,
                                                                (short)0, acc[q4 * 4 + 0], false, false);
      acc[q4 * 4 + 1] = __builtin_amdgcn_wmma_f32_16x16x32_bf16(false, a, false, b1,
                                                                (short)0, acc[q4 * 4 + 1], false, false);
      acc[q4 * 4 + 2] = __builtin_amdgcn_wmma_f32_16x16x32_bf16(false, a, false, b2,
                                                                (short)0, acc[q4 * 4 + 2], false, false);
      acc[q4 * 4 + 3] = __builtin_amdgcn_wmma_f32_16x16x32_bf16(false, a, false, b3,
                                                                (short)0, acc[q4 * 4 + 3], false, false);
    }

    // commit staged regs into the other LDS buffer; single barrier per chunk
    if (kc + 1 < 32) {
      char* ldst = (char*)(&bstage[(kc + 1) & 1][0]) + toff;
#pragma unroll
      for (int q = 0; q < 8; ++q) *(v4u*)(ldst + q * 16) = st[q];
    }
    __syncthreads();
  }

  // In-register fused epilogue: lane holds rows r+8*lhalf at column lrow for
  // every gate group (acc[kk*4+g]). 8 (b,h) pairs per lane.
  const int h = hb * 16 + lrow;
#pragma unroll
  for (int r = 0; r < 8; ++r) {
    const int b  = mbase + r + 8 * lhalf;
    const int xi = X[b * SEQ + t];
    const float* wiRow = Wi + (size_t)xi * GSZ;

    float bestO = -3.0e38f, cSel = 0.0f;
#pragma unroll
    for (int kk = 0; kk < KCELL; ++kk) {
      const int nb = kk * 4096 + h;            // n_global of gate 0
      float gi = acc[kk * 4 + 0][r] + wiRow[nb]        + bsum[nb];
      float gf = acc[kk * 4 + 1][r] + wiRow[nb + 1024] + bsum[nb + 1024];
      float gg = acc[kk * 4 + 2][r] + wiRow[nb + 2048] + bsum[nb + 2048];
      float go = acc[kk * 4 + 3][r] + wiRow[nb + 3072] + bsum[nb + 3072];
      const size_t ci = ((size_t)b * KCELL + kk) * HID + h;
      float cn = sigf(gf) * cSt[ci] + sigf(gi) * tanhf(gg);
      cSt[ci] = cn;
      if (go > bestO) { bestO = go; cSel = cn; }  // strict > keeps first-max (argmax tie rule)
    }
    const float hn = sigf(bestO) * tanhf(cSel);
    hOut[(size_t)b * HID + h] = (__bf16)hn;
  }
}

// ---------------- final projection: logits = h @ Wl + bl ----------------
__global__ __launch_bounds__(256) void logits_kernel(
    const __bf16* __restrict__ hB, const __bf16* __restrict__ WlF,
    const float* __restrict__ bl, float* __restrict__ out) {
  const int mb    = blockIdx.y * 32;
  const int wv    = threadIdx.x >> 5;
  const int lane  = threadIdx.x & 31;
  const int lrow  = lane & 15;
  const int lhalf = lane >> 4;
  const int mt    = wv >> 2;                    // 0..1
  const int ntg   = blockIdx.x * 4 + (wv & 3);  // 0..63

  v8f acc = {};
  const __bf16* hRow  = hB + (size_t)(mb + mt * 16 + lrow) * HID;
  const __bf16* bBase = WlF + ((size_t)ntg * 32 * 32 + lane) * 16;
  for (int kc = 0; kc < 32; ++kc) {
    const int kb = kc * 32 + lhalf * 16;
    v16bf a;
    *((v8bf*)&a)     = *(const v8bf*)(hRow + kb);
    *((v8bf*)&a + 1) = *(const v8bf*)(hRow + kb + 8);
    v16bf b = *(const v16bf*)(bBase + (size_t)kc * 512);
    acc = __builtin_amdgcn_wmma_f32_16x16x32_bf16(false, a, false, b,
                                                  (short)0, acc, false, false);
  }
  const int col = ntg * 16 + lrow;
  const float blv = bl[col];
#pragma unroll
  for (int r = 0; r < 8; ++r) {
    const int row = mb + mt * 16 + r + 8 * lhalf;
    out[(size_t)row * 1024 + col] = acc[r] + blv;
  }
}

// ---------------- in-place row log-softmax ----------------
__global__ __launch_bounds__(256) void lsm_kernel(float* __restrict__ out) {
  __shared__ float red[256];
  float* row = out + (size_t)blockIdx.x * 1024;
  float m = -3.0e38f;
  for (int i = threadIdx.x; i < 1024; i += 256) m = fmaxf(m, row[i]);
  red[threadIdx.x] = m;
  __syncthreads();
  for (int s = 128; s > 0; s >>= 1) {
    if (threadIdx.x < s) red[threadIdx.x] = fmaxf(red[threadIdx.x], red[threadIdx.x + s]);
    __syncthreads();
  }
  m = red[0];
  __syncthreads();
  float sum = 0.0f;
  for (int i = threadIdx.x; i < 1024; i += 256) sum += __expf(row[i] - m);
  red[threadIdx.x] = sum;
  __syncthreads();
  for (int s = 128; s > 0; s >>= 1) {
    if (threadIdx.x < s) red[threadIdx.x] += red[threadIdx.x + s];
    __syncthreads();
  }
  const float lse = m + __logf(red[0]);
  for (int i = threadIdx.x; i < 1024; i += 256) row[i] -= lse;
}

extern "C" void kernel_launch(void* const* d_in, const int* in_sizes, int n_in,
                              void* d_out, int out_size, void* d_ws, size_t ws_size,
                              hipStream_t stream) {
  (void)in_sizes; (void)n_in; (void)out_size; (void)ws_size;
  const int*   X  = (const int*)d_in[0];
  const float* Wi = (const float*)d_in[1];
  const float* bi = (const float*)d_in[2];
  const float* Wh = (const float*)d_in[3];
  const float* bh = (const float*)d_in[4];
  const float* Wl = (const float*)d_in[5];
  const float* bl = (const float*)d_in[6];
  float* out = (float*)d_out;

  char* ws = (char*)d_ws;
  __bf16* WhF  = (__bf16*)(ws + OFF_WHF);
  __bf16* WlF  = (__bf16*)(ws + OFF_WLF);
  __bf16* hB0  = (__bf16*)(ws + OFF_HB0);
  __bf16* hB1  = (__bf16*)(ws + OFF_HB1);
  float*  cSt  = (float*)(ws + OFF_C);
  float*  bsum = (float*)(ws + OFF_BS);

  // 1) pre-swizzle weights into WMMA fragment layouts (bf16); fuse bi+bh
  prep_whf<<<4096, 256, 0, stream>>>(Wh, WhF);
  prep_frags<<<256, 256, 0, stream>>>(Wl, WlF, 1024, 64);
  prep_bsum<<<GSZ / 256, 256, 0, stream>>>(bi, bh, bsum);

  // 2) zero recurrent state (h ping-pong buffers + c)
  zero_ws<<<(ZERO_BYTES / 4 + 255) / 256, 256, 0, stream>>>((unsigned int*)(ws + OFF_HB0),
                                                            ZERO_BYTES / 4);

  // 3) 128 fused recurrent steps, ping-pong bf16 h buffers
  __bf16* bufs[2] = {hB0, hB1};
  for (int t = 0; t < SEQ; ++t) {
    lstm_step<<<dim3(64, 4), 128, 0, stream>>>(bufs[t & 1], bufs[(t + 1) & 1], cSt,
                                               WhF, Wi, bsum, X, t);
  }

  // 4) final projection (last h is in bufs[SEQ & 1] == hB0) + log-softmax
  logits_kernel<<<dim3(16, 8), 256, 0, stream>>>(bufs[SEQ & 1], WlF, bl, out);
  lsm_kernel<<<BATCH, 256, 0, stream>>>(out);
}
